// up_block_57458072486023
// MI455X (gfx1250) — matrix operand order repaired and checked by reading, not compiled, not run on hardware
//
#include <hip/hip_runtime.h>
#include <hip/hip_bf16.h>

typedef __attribute__((ext_vector_type(16))) __bf16 v16bf;
typedef __attribute__((ext_vector_type(8)))  float  v8f;

#define B_   4
#define C_   128
#define N_   1024
#define KNN_ 20
#define M_   4096      /* N * UP */
#define KP_  160       /* 130 channels padded to 5*32 */
#define EPS_ 1e-5f
#define NEGBIG_ -3.0e38f

// ---------------------------------------------------------------------------
// WMMA fragment helpers (ISA 16-bit A / B / 32-bit C-D layouts, wave32)
// ---------------------------------------------------------------------------
static __device__ __forceinline__ v16bf load_a_bf16(const __bf16* src, int ld, int lane) {
  // A 16(M)x32(K): lane holds row (lane&15); h=lane>>4 selects K ranges
  int r = lane & 15, h = lane >> 4;
  const __bf16* p = src + (size_t)r * ld;
  v16bf a;
#pragma unroll
  for (int i = 0; i < 8; ++i) a[i] = p[h * 8 + i];
#pragma unroll
  for (int i = 0; i < 8; ++i) a[8 + i] = p[16 + h * 8 + i];
  return a;
}

static __device__ __forceinline__ v16bf load_b_bf16(const __bf16* src, int ld, int lane) {
  // B 32(K)x16(N): lane holds column (lane&15); h=lane>>4 selects K ranges (same as A).
  // src must point at the tile such that column c's K-vector is contiguous
  // at src + c*ld.
  int c = lane & 15, h = lane >> 4;
  const __bf16* p = src + (size_t)c * ld;
  v16bf b;
#pragma unroll
  for (int i = 0; i < 8; ++i) b[i] = p[h * 8 + i];
#pragma unroll
  for (int i = 0; i < 8; ++i) b[8 + i] = p[16 + h * 8 + i];
  return b;
}

static __device__ __forceinline__ v8f wmma_bf16(v16bf a, v16bf b, v8f c) {
  return __builtin_amdgcn_wmma_f32_16x16x32_bf16(false, a, false, b, (short)0, c,
                                                 false, false);
}

// ---------------------------------------------------------------------------
// 0) weight packing (fp32 -> bf16, with channel padding to multiples of 16/32)
// ---------------------------------------------------------------------------
__global__ void pack_weights_kernel(const float* Wec, const float* Wf, const float* Wg,
                                    const float* Wh, const float* W1, const float* W2,
                                    __bf16* dWec, __bf16* dWf, __bf16* dWg,
                                    __bf16* dWh, __bf16* dW1, __bf16* dW2) {
  int tid = blockIdx.x * blockDim.x + threadIdx.x;
  int nt  = gridDim.x * blockDim.x;
  for (int i = tid; i < 512 * 256; i += nt) dWec[i] = (__bf16)Wec[i];
  for (int i = tid; i < 32 * KP_; i += nt) {
    int r = i / KP_, c = i % KP_;
    dWf[i] = (__bf16)(c < 130 ? Wf[r * 130 + c] : 0.f);
    dWg[i] = (__bf16)(c < 130 ? Wg[r * 130 + c] : 0.f);
  }
  for (int i = tid; i < 144 * KP_; i += nt) {
    int r = i / KP_, c = i % KP_;
    dWh[i] = (__bf16)((r < 130 && c < 130) ? Wh[r * 130 + c] : 0.f);
  }
  for (int i = tid; i < 256 * KP_; i += nt) {
    int r = i / KP_, c = i % KP_;
    dW1[i] = (__bf16)(c < 130 ? W1[r * 130 + c] : 0.f);
  }
  for (int i = tid; i < 128 * 256; i += nt) dW2[i] = (__bf16)W2[i];
}

// ---------------------------------------------------------------------------
// 1) x -> point-major fp32/bf16 copies + squared norms
// ---------------------------------------------------------------------------
__global__ void prep_x_kernel(const float* x, float* xT, __bf16* xTbf, float* sq) {
  int t = blockIdx.x * blockDim.x + threadIdx.x;
  if (t >= B_ * N_) return;
  int b = t / N_, n = t % N_;
  float acc = 0.f;
  for (int c = 0; c < C_; ++c) {
    float v = x[((size_t)b * C_ + c) * N_ + n];
    xT[(size_t)t * C_ + c]   = v;
    xTbf[(size_t)t * C_ + c] = (__bf16)v;
    acc += v * v;
  }
  sq[t] = acc;
}

__global__ void init_kernel(float* net1raw, float* ec_sum, float* ec_sqs) {
  int t  = blockIdx.x * blockDim.x + threadIdx.x;
  int nt = gridDim.x * blockDim.x;
  for (int i = t; i < B_ * 512 * N_; i += nt) net1raw[i] = NEGBIG_;
  for (int i = t; i < 512; i += nt) { ec_sum[i] = 0.f; ec_sqs[i] = 0.f; }
}

// ---------------------------------------------------------------------------
// 2) Gram / negative-distance matrix via WMMA (one wave per 16x16 tile)
// ---------------------------------------------------------------------------
__global__ void gram_kernel(const __bf16* xTbf, const float* sq, float* negd) {
  int wave = threadIdx.x >> 5, lane = threadIdx.x & 31;
  int t  = blockIdx.x * 4 + wave;
  int b  = t >> 12;                // 64*64 tiles per batch
  int r  = t & 4095;
  int mt = r >> 6, nt = r & 63;
  const __bf16* base = xTbf + (size_t)b * N_ * C_;
  v8f acc = {0.f, 0.f, 0.f, 0.f, 0.f, 0.f, 0.f, 0.f};
#pragma unroll
  for (int k = 0; k < C_; k += 32) {
    v16bf A  = load_a_bf16(base + (size_t)(mt * 16) * C_ + k, C_, lane);
    v16bf Bm = load_b_bf16(base + (size_t)(nt * 16) * C_ + k, C_, lane);
    acc = wmma_bf16(A, Bm, acc);
  }
  int h = lane >> 4, col = lane & 15;
  int n = nt * 16 + col;
  float sqn = sq[b * N_ + n];
#pragma unroll
  for (int v = 0; v < 8; ++v) {
    int m = mt * 16 + h * 8 + v;
    negd[((size_t)(b * N_ + m)) * N_ + n] = 2.f * acc[v] - sq[b * N_ + m] - sqn;
  }
}

// ---------------------------------------------------------------------------
// 3) top-20 per row (one wave per row, iterative wave-wide argmax)
// ---------------------------------------------------------------------------
__global__ void topk_kernel(const float* negd, int* idx) {
  int wave = threadIdx.x >> 5, lane = threadIdx.x & 31;
  int row  = blockIdx.x * 4 + wave;
  const float* d = negd + (size_t)row * N_;
  float vals[32];
#pragma unroll
  for (int i = 0; i < 32; ++i) vals[i] = d[lane + i * 32];
  int* out = idx + (size_t)row * KNN_;
  for (int s = 0; s < KNN_; ++s) {
    float best = NEGBIG_; int bi = 0x7fffffff;
#pragma unroll
    for (int i = 0; i < 32; ++i)
      if (vals[i] > best) { best = vals[i]; bi = lane + i * 32; }
#pragma unroll
    for (int msk = 16; msk >= 1; msk >>= 1) {
      float ov = __shfl_xor(best, msk, 32);
      int   oi = __shfl_xor(bi, msk, 32);
      if (ov > best || (ov == best && oi < bi)) { best = ov; bi = oi; }
    }
    if (lane == 0) out[s] = bi;
#pragma unroll
    for (int i = 0; i < 32; ++i)
      if (lane + i * 32 == bi) vals[i] = NEGBIG_;
  }
}

// ---------------------------------------------------------------------------
// 4) edge-conv GEMM (single pass): LDS gather of 16 edge rows x 256 channels,
//    WMMA against W_ec, fused per-channel sum/sumsq stats (atomicAdd) and
//    max-over-k (global f32 atomic max on raw outputs; BN+leaky are monotone).
// ---------------------------------------------------------------------------
__global__ void edgeconv_kernel(const float* xT, const int* idx, const __bf16* Wec,
                                float* net1raw, float* ec_sum, float* ec_sqs) {
  __shared__ __bf16 eds[16][264];   // 16 edge rows x 256 channels (+8 pad)
  int b   = blockIdx.x / 1280;
  int nk0 = (blockIdx.x % 1280) * 16;
  int tid = threadIdx.x;
  for (int e = tid; e < 16 * 256; e += 128) {
    int row = e >> 8, c = e & 255;
    int nkg = nk0 + row;
    int n = nkg / KNN_, kk = nkg % KNN_;
    int j = idx[(b * N_ + n) * KNN_ + kk];
    float val;
    if (c < 128)
      val = xT[((size_t)(b * N_ + j)) * C_ + c] - xT[((size_t)(b * N_ + n)) * C_ + c];
    else
      val = xT[((size_t)(b * N_ + n)) * C_ + (c - 128)];
    eds[row][c] = (__bf16)val;
  }
  __syncthreads();
  int wave = tid >> 5, lane = tid & 31;
  int h = lane >> 4, col = lane & 15;
  int nkg = nk0 + col;
  int n   = nkg / KNN_;
  for (int ot = wave; ot < 32; ot += 4) {
    v8f acc = {0.f, 0.f, 0.f, 0.f, 0.f, 0.f, 0.f, 0.f};
#pragma unroll
    for (int k = 0; k < 256; k += 32) {
      v16bf A  = load_a_bf16(Wec + (size_t)(ot * 16) * 256 + k, 256, lane);
      v16bf Bm = load_b_bf16(&eds[0][0] + k, 264, lane);
      acc = wmma_bf16(A, Bm, acc);
    }
#pragma unroll
    for (int v = 0; v < 8; ++v) {
      int o = ot * 16 + h * 8 + v;
      float val = acc[v];
      float s = val, s2 = val * val;
#pragma unroll
      for (int msk = 8; msk >= 1; msk >>= 1) {
        s  += __shfl_xor(s, msk, 32);
        s2 += __shfl_xor(s2, msk, 32);
      }
      if (col == 0) { atomicAdd(&ec_sum[o], s); atomicAdd(&ec_sqs[o], s2); }
      __hip_atomic_fetch_max(&net1raw[((size_t)(b * 512 + o)) * N_ + n], val,
                             __ATOMIC_RELAXED, __HIP_MEMORY_SCOPE_AGENT);
    }
  }
}

__global__ void ec_finalize_kernel(const float* sum, const float* sqs,
                                   float* mean, float* inv) {
  int o = blockIdx.x * blockDim.x + threadIdx.x;
  if (o < 512) {
    float cnt = (float)(B_ * N_ * KNN_);
    float mu  = sum[o] / cnt;
    float var = sqs[o] / cnt - mu * mu;
    mean[o] = mu;
    inv[o]  = rsqrtf(var + EPS_);
  }
}

// ---------------------------------------------------------------------------
// 5) BN + leaky on the maxed raw values, reshape to (B,130,M), append grid,
//    and emit the bf16 point-major (B, M, 160) copy for the conv GEMMs.
// ---------------------------------------------------------------------------
__global__ void build_net_kernel(const float* net1raw, const float* mean, const float* inv,
                                 const float* gec, const float* bec,
                                 float* net130, __bf16* netT) {
  int t = blockIdx.x * blockDim.x + threadIdx.x;
  if (t >= B_ * M_) return;
  int b = t / M_, m = t % M_;
  int n = m >> 2, u = m & 3;
  for (int c = 0; c < 128; ++c) {
    int o = u * 128 + c;
    float raw = net1raw[((size_t)(b * 512 + o)) * N_ + n];
    float v = (raw - mean[o]) * inv[o] * gec[o] + bec[o];
    v = v > 0.f ? v : 0.2f * v;
    net130[((size_t)(b * 130 + c)) * M_ + m] = v;
    netT[(size_t)t * KP_ + c] = (__bf16)v;
  }
  int gi = m >> 10;
  float gx = (gi >= 2) ? 0.2f : -0.2f;
  float gy = (gi & 1) ? 0.2f : -0.2f;
  net130[((size_t)(b * 130 + 128)) * M_ + m] = gx;
  net130[((size_t)(b * 130 + 129)) * M_ + m] = gy;
  netT[(size_t)t * KP_ + 128] = (__bf16)gx;
  netT[(size_t)t * KP_ + 129] = (__bf16)gy;
  for (int c = 130; c < KP_; ++c) netT[(size_t)t * KP_ + c] = (__bf16)0.f;
}

// ---------------------------------------------------------------------------
// 6) generic 1x1-conv GEMM: out(B,CoPad,M) = W(CoPad,Kpad) @ inT(B,M,Kpad)^T
//    one wave per 16x16 (o,m) tile; optional bias+relu, f32 and/or bf16T out.
// ---------------------------------------------------------------------------
__global__ void conv_gemm_kernel(const __bf16* inT, const __bf16* W, const float* bias,
                                 float* outF, __bf16* outTbf,
                                 int Kpad, int CoPad, int relu) {
  int wave = threadIdx.x >> 5, lane = threadIdx.x & 31;
  int t = blockIdx.x * 4 + wave;
  int perB = (CoPad >> 4) * (M_ >> 4);
  int b  = t / perB;
  int r  = t % perB;
  int ot = r / (M_ >> 4);
  int mt = r % (M_ >> 4);
  const __bf16* ibase = inT + ((size_t)b * M_ + mt * 16) * Kpad;
  const __bf16* wbase = W + (size_t)(ot * 16) * Kpad;
  v8f acc = {0.f, 0.f, 0.f, 0.f, 0.f, 0.f, 0.f, 0.f};
  for (int k = 0; k < Kpad; k += 32) {
    v16bf A  = load_a_bf16(wbase + k, Kpad, lane);
    v16bf Bm = load_b_bf16(ibase + k, Kpad, lane);
    acc = wmma_bf16(A, Bm, acc);
  }
  int h = lane >> 4, col = lane & 15;
  int m = mt * 16 + col;
#pragma unroll
  for (int v = 0; v < 8; ++v) {
    int o = ot * 16 + h * 8 + v;
    float val = acc[v] + (bias ? bias[o] : 0.f);
    if (relu) val = fmaxf(val, 0.f);
    if (outF)   outF[((size_t)b * CoPad + o) * M_ + m] = val;
    if (outTbf) outTbf[((size_t)b * M_ + m) * CoPad + o] = (__bf16)val;
  }
}

// ---------------------------------------------------------------------------
// 7) per-channel BN stats over (B, M)
// ---------------------------------------------------------------------------
__global__ void stats_kernel(const float* y, int CoPad, float* mean, float* inv) {
  int c = blockIdx.x;
  __shared__ float s1[256], s2[256];
  float a = 0.f, b2 = 0.f;
  for (int i = threadIdx.x; i < B_ * M_; i += blockDim.x) {
    int bb = i / M_, m = i % M_;
    float v = y[((size_t)bb * CoPad + c) * M_ + m];
    a += v; b2 += v * v;
  }
  s1[threadIdx.x] = a; s2[threadIdx.x] = b2;
  __syncthreads();
  for (int s = 128; s > 0; s >>= 1) {
    if ((int)threadIdx.x < s) {
      s1[threadIdx.x] += s1[threadIdx.x + s];
      s2[threadIdx.x] += s2[threadIdx.x + s];
    }
    __syncthreads();
  }
  if (threadIdx.x == 0) {
    float cnt = (float)(B_ * M_);
    float mu  = s1[0] / cnt;
    float var = s2[0] / cnt - mu * mu;
    mean[c] = mu;
    inv[c]  = rsqrtf(var + EPS_);
  }
}

// ---------------------------------------------------------------------------
// 8) BN+relu apply -> attention operand layouts:
//    gQT,fKT: (B, M, 32) bf16 (K-contiguous)   hV: (B, 144, M) bf16 (M-contig)
// ---------------------------------------------------------------------------
__global__ void bn_apply_kernel(const float* yf, const float* yg, const float* yh,
                                const float* mf, const float* invf, const float* gf, const float* betf,
                                const float* mg, const float* invg, const float* gg, const float* betg,
                                const float* mh, const float* invh, const float* gh, const float* beth,
                                __bf16* gQT, __bf16* fKT, __bf16* hV) {
  int t = blockIdx.x * blockDim.x + threadIdx.x;
  if (t >= B_ * M_) return;
  int b = t / M_, m = t % M_;
  for (int c = 0; c < 32; ++c) {
    float fv = (yf[((size_t)(b * 32 + c)) * M_ + m] - mf[c]) * invf[c] * gf[c] + betf[c];
    fKT[(size_t)t * 32 + c] = (__bf16)fmaxf(fv, 0.f);
    float gv = (yg[((size_t)(b * 32 + c)) * M_ + m] - mg[c]) * invg[c] * gg[c] + betg[c];
    gQT[(size_t)t * 32 + c] = (__bf16)fmaxf(gv, 0.f);
  }
  for (int c = 0; c < 144; ++c) {
    float hv = 0.f;
    if (c < 130)
      hv = fmaxf((yh[((size_t)(b * 144 + c)) * M_ + m] - mh[c]) * invh[c] * gh[c] + beth[c], 0.f);
    hV[((size_t)(b * 144 + c)) * M_ + m] = (__bf16)hv;
  }
}

// ---------------------------------------------------------------------------
// 9) flash-style attention: one wave per 16-query tile; QK is a single bf16
//    WMMA per 16-key tile (K = head dim 32), online softmax in f32, P routed
//    through LDS to A-layout, PV = 9 WMMAs per 32-key step (144 value ch).
// ---------------------------------------------------------------------------
__global__ void flash_attn_kernel(const __bf16* gQT, const __bf16* fKT,
                                  const __bf16* hV, float* oatt) {
  __shared__ __bf16 pshare[4][16][32];
  int wave = threadIdx.x >> 5, lane = threadIdx.x & 31;
  int t  = blockIdx.x * 4 + wave;
  int b  = t >> 8, mt = t & 255;
  int h  = lane >> 4, col = lane & 15;
  v16bf Aq = load_a_bf16(gQT + ((size_t)b * M_ + mt * 16) * 32, 32, lane);
  v8f zero = {0.f, 0.f, 0.f, 0.f, 0.f, 0.f, 0.f, 0.f};
  v8f O[9];
#pragma unroll
  for (int i = 0; i < 9; ++i) O[i] = zero;
  float mrow[8], lrow[8], sc[8];
#pragma unroll
  for (int v = 0; v < 8; ++v) { mrow[v] = NEGBIG_; lrow[v] = 0.f; }
  const __bf16* fb = fKT + (size_t)b * M_ * 32;
  const __bf16* vb = hV + (size_t)b * 144 * M_;
  for (int nb = 0; nb < M_; nb += 32) {
    v16bf K0 = load_b_bf16(fb + (size_t)nb * 32, 32, lane);
    v16bf K1 = load_b_bf16(fb + (size_t)(nb + 16) * 32, 32, lane);
    v8f S0 = wmma_bf16(Aq, K0, zero);
    v8f S1 = wmma_bf16(Aq, K1, zero);
#pragma unroll
    for (int v = 0; v < 8; ++v) {
      float mx = fmaxf(S0[v], S1[v]);
#pragma unroll
      for (int msk = 8; msk >= 1; msk >>= 1) mx = fmaxf(mx, __shfl_xor(mx, msk, 32));
      float mn = fmaxf(mrow[v], mx);
      sc[v] = __expf(mrow[v] - mn);
      float p0 = __expf(S0[v] - mn);
      float p1 = __expf(S1[v] - mn);
      float rs = p0 + p1;
#pragma unroll
      for (int msk = 8; msk >= 1; msk >>= 1) rs += __shfl_xor(rs, msk, 32);
      lrow[v] = lrow[v] * sc[v] + rs;
      mrow[v] = mn;
      pshare[wave][h * 8 + v][col]      = (__bf16)p0;
      pshare[wave][h * 8 + v][16 + col] = (__bf16)p1;
    }
#pragma unroll
    for (int i = 0; i < 9; ++i)
#pragma unroll
      for (int v = 0; v < 8; ++v) O[i][v] *= sc[v];
    __syncthreads();
    v16bf Ap = load_a_bf16(&pshare[wave][0][0], 32, lane);
#pragma unroll
    for (int i = 0; i < 9; ++i) {
      v16bf Vb = load_b_bf16(vb + (size_t)(i * 16) * M_ + nb, M_, lane);
      O[i] = wmma_bf16(Ap, Vb, O[i]);
    }
    __syncthreads();
  }
#pragma unroll
  for (int v = 0; v < 8; ++v) {
    int m = mt * 16 + h * 8 + v;
    float invl = 1.f / lrow[v];
#pragma unroll
    for (int i = 0; i < 9; ++i) {
      int c = i * 16 + col;
      oatt[((size_t)(b * 144 + c)) * M_ + m] = O[i][v] * invl;
    }
  }
}

// ---------------------------------------------------------------------------
// 10) residual: gamma*o + net, emit bf16 point-major padded input for W1 GEMM
// ---------------------------------------------------------------------------
__global__ void residual_kernel(const float* oatt, const float* net130,
                                const float* gamma, __bf16* netattT) {
  int t = blockIdx.x * blockDim.x + threadIdx.x;
  if (t >= B_ * M_) return;
  int b = t / M_, m = t % M_;
  float ga = gamma[0];
  for (int c = 0; c < 130; ++c) {
    float val = ga * oatt[((size_t)(b * 144 + c)) * M_ + m] +
                net130[((size_t)(b * 130 + c)) * M_ + m];
    netattT[(size_t)t * KP_ + c] = (__bf16)val;
  }
  for (int c = 130; c < KP_; ++c) netattT[(size_t)t * KP_ + c] = (__bf16)0.f;
}

// ---------------------------------------------------------------------------
extern "C" void kernel_launch(void* const* d_in, const int* in_sizes, int n_in,
                              void* d_out, int out_size, void* d_ws, size_t ws_size,
                              hipStream_t stream) {
  const float* x    = (const float*)d_in[0];
  const float* Wec  = (const float*)d_in[1];
  const float* gec  = (const float*)d_in[2];
  const float* bec  = (const float*)d_in[3];
  const float* Wf   = (const float*)d_in[4];
  const float* gf   = (const float*)d_in[6];
  const float* betf = (const float*)d_in[7];
  const float* Wg   = (const float*)d_in[8];
  const float* gg   = (const float*)d_in[10];
  const float* betg = (const float*)d_in[11];
  const float* Wh   = (const float*)d_in[12];
  const float* gh   = (const float*)d_in[14];
  const float* beth = (const float*)d_in[15];
  const float* gamma= (const float*)d_in[16];
  const float* W1   = (const float*)d_in[17];
  const float* b1   = (const float*)d_in[18];
  const float* W2   = (const float*)d_in[19];
  const float* b2   = (const float*)d_in[20];
  (void)in_sizes; (void)n_in; (void)out_size; (void)ws_size;
  // biases bf/bg/bh precede BN => mathematically irrelevant; skipped.

  char* base = (char*)d_ws;
  size_t off = 0;
  auto alloc = [&](size_t bytes) -> void* {
    void* p = base + off;
    off = (off + bytes + 255) & ~(size_t)255;
    return p;
  };
  float*  xT      = (float*) alloc((size_t)B_ * N_ * C_ * 4);
  __bf16* xTbf    = (__bf16*)alloc((size_t)B_ * N_ * C_ * 2);
  float*  sq      = (float*) alloc((size_t)B_ * N_ * 4);
  float*  negd    = (float*) alloc((size_t)B_ * N_ * N_ * 4);
  int*    idx     = (int*)   alloc((size_t)B_ * N_ * KNN_ * 4);
  __bf16* dWec    = (__bf16*)alloc(512 * 256 * 2);
  __bf16* dWf     = (__bf16*)alloc(32 * KP_ * 2);
  __bf16* dWg     = (__bf16*)alloc(32 * KP_ * 2);
  __bf16* dWh     = (__bf16*)alloc(144 * KP_ * 2);
  __bf16* dW1     = (__bf16*)alloc(256 * KP_ * 2);
  __bf16* dW2     = (__bf16*)alloc(128 * 256 * 2);
  float*  ec_sum  = (float*) alloc(512 * 4);
  float*  ec_sqs  = (float*) alloc(512 * 4);
  float*  ec_mean = (float*) alloc(512 * 4);
  float*  ec_inv  = (float*) alloc(512 * 4);
  float*  net1raw = (float*) alloc((size_t)B_ * 512 * N_ * 4);
  float*  net130  = (float*) alloc((size_t)B_ * 130 * M_ * 4);
  __bf16* netT    = (__bf16*)alloc((size_t)B_ * M_ * KP_ * 2);
  float*  yh      = (float*) alloc((size_t)B_ * 144 * M_ * 4);
  float*  stf_m   = (float*) alloc(32 * 4);
  float*  stf_i   = (float*) alloc(32 * 4);
  float*  stg_m   = (float*) alloc(32 * 4);
  float*  stg_i   = (float*) alloc(32 * 4);
  float*  sth_m   = (float*) alloc(144 * 4);
  float*  sth_i   = (float*) alloc(144 * 4);
  __bf16* hV      = (__bf16*)alloc((size_t)B_ * 144 * M_ * 2);
  __bf16* natT    = (__bf16*)alloc((size_t)B_ * M_ * KP_ * 2);

  // Aliased regions (lifetimes disjoint):
  float*  yf   = xT;                              // xT dead after edgeconv
  __bf16* gQT  = xTbf;                            // xTbf dead after gram
  float*  oatt = negd;                            // negd dead after topk
  __bf16* fKT  = (__bf16*)((char*)negd + (size_t)B_ * 144 * M_ * 4);
  float*  yg   = (float*)((char*)negd + (size_t)B_ * 144 * M_ * 4 +
                          (size_t)B_ * M_ * 32 * 2);
  __bf16* y1T  = (__bf16*)net1raw;                // net1raw dead after build_net

  pack_weights_kernel<<<64, 256, 0, stream>>>(Wec, Wf, Wg, Wh, W1, W2,
                                              dWec, dWf, dWg, dWh, dW1, dW2);
  prep_x_kernel<<<(B_ * N_ + 255) / 256, 256, 0, stream>>>(x, xT, xTbf, sq);
  init_kernel<<<512, 256, 0, stream>>>(net1raw, ec_sum, ec_sqs);
  gram_kernel<<<4096, 128, 0, stream>>>(xTbf, sq, negd);
  topk_kernel<<<1024, 128, 0, stream>>>(negd, idx);
  edgeconv_kernel<<<5120, 128, 0, stream>>>(xT, idx, dWec, net1raw, ec_sum, ec_sqs);
  ec_finalize_kernel<<<2, 256, 0, stream>>>(ec_sum, ec_sqs, ec_mean, ec_inv);
  build_net_kernel<<<(B_ * M_ + 255) / 256, 256, 0, stream>>>(net1raw, ec_mean, ec_inv,
                                                              gec, bec, net130, netT);
  conv_gemm_kernel<<<(4 * 2 * 256) / 4, 128, 0, stream>>>(netT, dWf, nullptr, yf, nullptr, KP_, 32, 0);
  conv_gemm_kernel<<<(4 * 2 * 256) / 4, 128, 0, stream>>>(netT, dWg, nullptr, yg, nullptr, KP_, 32, 0);
  conv_gemm_kernel<<<(4 * 9 * 256) / 4, 128, 0, stream>>>(netT, dWh, nullptr, yh, nullptr, KP_, 144, 0);
  stats_kernel<<<32, 256, 0, stream>>>(yf, 32, stf_m, stf_i);
  stats_kernel<<<32, 256, 0, stream>>>(yg, 32, stg_m, stg_i);
  stats_kernel<<<130, 256, 0, stream>>>(yh, 144, sth_m, sth_i);
  bn_apply_kernel<<<(B_ * M_ + 255) / 256, 256, 0, stream>>>(
      yf, yg, yh, stf_m, stf_i, gf, betf, stg_m, stg_i, gg, betg,
      sth_m, sth_i, gh, beth, gQT, fKT, hV);
  flash_attn_kernel<<<256, 128, 0, stream>>>(gQT, fKT, hV, oatt);
  residual_kernel<<<(B_ * M_ + 255) / 256, 256, 0, stream>>>(oatt, net130, gamma, natT);
  conv_gemm_kernel<<<(4 * 16 * 256) / 4, 128, 0, stream>>>(natT, dW1, b1, nullptr, y1T, KP_, 256, 1);
  conv_gemm_kernel<<<(4 * 8 * 256) / 4, 128, 0, stream>>>(y1T, dW2, b2, (float*)d_out, nullptr, 256, 128, 1);
}
